// LMU_3865470566927
// MI455X (gfx1250) — matrix operand order repaired and checked by reading, not compiled, not run on hardware
//
#include <hip/hip_runtime.h>
#include <hip/hip_bf16.h>
#include <math.h>

// ---------------------------------------------------------------------------
// LMU scan on gfx1250: bf16 WMMA (v_wmma_f32_16x16x32_bf16), wave32.
// Sizes fixed by the reference: B=16, T=4096, Din=128, d=units=256.
// Scan structure: per step, update state (f32 kept in registers), publish bf16
// state to LDS once, then ONE pass over the state feeds TWO interleaved WMMA
// chains: y_t = tanh(S@QW + b) and next step's S@A + xB_{t+1}.
// ---------------------------------------------------------------------------

typedef __attribute__((ext_vector_type(16))) __bf16 v16bf;
typedef __attribute__((ext_vector_type(8)))  __bf16 v8bf;
typedef __attribute__((ext_vector_type(8)))  float  v8f;

#define TT    4096
#define BB    16
#define DIN   128
#define DD    256
#define ALPHA (1.0f - 1.0f / 128.0f)
#define BETA  (1.0f / 128.0f)

// --- prep: transpose+convert A (256x256) -> At[n][k] bf16, Bm (128x256) -> Bt[n][k] bf16
__global__ void lmu_prep_transpose(const float* __restrict__ A,
                                   const float* __restrict__ Bm,
                                   __bf16* __restrict__ At,
                                   __bf16* __restrict__ Bt) {
    int i = blockIdx.x * 256 + threadIdx.x;           // 0..65535
    {
        int n = i >> 8, k = i & 255;
        At[i] = (__bf16)A[k * DD + n];                // At[n][k] = A[k][n]
    }
    if (i < DD * DIN) {                               // 32768
        int n = i >> 7, k = i & 127;
        Bt[i] = (__bf16)Bm[k * DD + n];               // Bt[n][k] = Bm[k][n]
    }
}

// --- prep: QWt[n*256+k] = (Q @ W_mix)[k][n] in bf16.
// Q[k][m] = (2k+1)/16 if (k>m && (k+m) odd) else 0  (sqrt(d)=16)
__global__ void lmu_prep_qw(const float* __restrict__ Wmix,
                            __bf16* __restrict__ QWt) {
    int i = blockIdx.x * 256 + threadIdx.x;           // 0..65535
    int n = i >> 8, k = i & 255;
    float s = 0.0f;
    int m0 = (k & 1) ? 0 : 1;                         // parity(m) != parity(k)
    for (int m = m0; m < k; m += 2)
        s += Wmix[m * DD + n];
    s *= (float)(2 * k + 1) * (1.0f / 16.0f);
    QWt[n * DD + k] = (__bf16)s;
}

// --- phase 1: xB[t][b][n] = sum_i x[b][t][i] * Bm[i][n] via bf16 WMMA
// one wave per (t_tile, b, n_tile); 8 waves / block
__global__ void lmu_xb_gemm(const float* __restrict__ x,
                            const __bf16* __restrict__ Bt,
                            float* __restrict__ xB) {
    int w     = blockIdx.x * 8 + (threadIdx.x >> 5);  // global wave id, 0..65535
    int lane  = threadIdx.x & 31;
    int lrow  = lane & 15;
    bool hi   = lane >= 16;
    int ntile = w & 15;
    int b     = (w >> 4) & 15;
    int ttile = w >> 8;                               // 0..255
    int t_a   = ttile * 16 + lrow;                    // A-frag row (M = lane%16)
    int ncol  = ntile * 16 + lrow;                    // B/C column

    v8f c;
#pragma unroll
    for (int k = 0; k < 8; ++k) c[k] = 0.0f;

#pragma unroll
    for (int kc = 0; kc < 4; ++kc) {
        const float* xp = x + ((size_t)b * TT + t_a) * DIN + kc * 32 + (hi ? 16 : 0);
        v16bf a;
#pragma unroll
        for (int e = 0; e < 8; ++e) {
            a[e]     = (__bf16)xp[e];                 // K = kc*32 + hi*16 + e
            a[e + 8] = (__bf16)xp[e + 8];             // K = kc*32 + hi*16 + 8 + e
        }
        v16bf bb = *(const v16bf*)(Bt + ncol * DIN + kc * 32 + (hi ? 16 : 0));
        c = __builtin_amdgcn_wmma_f32_16x16x32_bf16(false, a, false, bb,
                                                    (short)0, c, false, false);
    }
#pragma unroll
    for (int k = 0; k < 8; ++k) {
        int M = k + (hi ? 8 : 0);
        int t = ttile * 16 + M;
        xB[((size_t)t * BB + b) * DD + ncol] = c[k];
    }
}

// A-fragment (16x32 bf16) from LDS state copy, in contiguous k order (element e <-> K = kc*32 + 16*hi + e)
__device__ __forceinline__ v16bf lmu_load_afrag(const __bf16* Sbf, int lrow,
                                                bool hi, int kc) {
    const __bf16* p = Sbf + lrow * DD + kc * 32 + (hi ? 16 : 0);
    v8bf lo = *(const v8bf*)(p);                      // K = base .. base+7
    v8bf ho = *(const v8bf*)(p + 8);                  // K = base+8 .. base+15
    v16bf a;
#pragma unroll
    for (int e = 0; e < 8; ++e) { a[e] = lo[e]; a[e + 8] = ho[e]; }
    return a;
}

// --- phase 2: sequential scan. 1 block, 16 waves; wave j owns columns 16j..16j+15.
// A and QW column slices preloaded into registers (8 v16bf B-frags each).
// f32 state lives entirely in registers (each lane updates exactly its own
// C-fragment positions); only the bf16 WMMA operand copy goes through LDS.
__global__ __launch_bounds__(512, 1)
void lmu_scan(const float* __restrict__ xB,
              const __bf16* __restrict__ At,
              const __bf16* __restrict__ QWt,
              const float* __restrict__ bmix,
              float* __restrict__ out) {
    __shared__ __bf16 Sbf[BB * DD];                   // bf16 state, WMMA A-operand

    int tid  = threadIdx.x;
    int wave = tid >> 5;                              // 0..15
    int lane = tid & 31;
    int lrow = lane & 15;
    bool hi  = lane >= 16;
    int ncol = wave * 16 + lrow;                      // global column 0..255

    // Preload resident B-fragments for A and QW (this wave's 16 columns).
    v16bf bA[8], bQ[8];
#pragma unroll
    for (int kc = 0; kc < 8; ++kc) {
        int kb = kc * 32 + (hi ? 16 : 0);
        bA[kc] = *(const v16bf*)(At  + ncol * DD + kb);
        bQ[kc] = *(const v16bf*)(QWt + ncol * DD + kb);
    }
    float bmv = bmix[ncol];

    // f32 state in registers: sreg[k] = S[k + 8*hi][ncol]; S_0 = 0.
    v8f sreg;
#pragma unroll
    for (int k = 0; k < 8; ++k) sreg[k] = 0.0f;

    // Prologue: c = S_0@A + xB_0 = xB_0 tile.
    v8f c;
#pragma unroll
    for (int k = 0; k < 8; ++k) {
        int M = k + (hi ? 8 : 0);
        c[k] = xB[M * DD + ncol];
    }

    for (int t = 0; t < TT; ++t) {
        // S_{t+1} = alpha*S_t + beta*(S_t@A + xB_t); publish bf16 copy.
#pragma unroll
        for (int k = 0; k < 8; ++k) {
            int M = k + (hi ? 8 : 0);
            float sn = ALPHA * sreg[k] + BETA * c[k];
            sreg[k] = sn;
            Sbf[M * DD + ncol] = (__bf16)sn;
        }
        __syncthreads();                              // S_{t+1} visible to all

        int tn = (t + 1 < TT) ? t + 1 : TT - 1;       // clamped: last result unused
        const float* xbn = xB + (size_t)tn * (BB * DD);
        if (t + 2 < TT)
            __builtin_prefetch(xbn + (BB * DD) + (hi ? 8 * DD : 0) + ncol, 0, 1);

        // One pass over S_{t+1}: two interleaved WMMA chains.
        v8f cy, cn;
#pragma unroll
        for (int k = 0; k < 8; ++k) {
            int M = k + (hi ? 8 : 0);
            cy[k] = bmv;                              // y  = S@QW + b_mix
            cn[k] = xbn[M * DD + ncol];               // c' = S@A + xB_{t+1}
        }
#pragma unroll
        for (int kc = 0; kc < 8; ++kc) {
            v16bf a = lmu_load_afrag(Sbf, lrow, hi, kc);
            cy = __builtin_amdgcn_wmma_f32_16x16x32_bf16(false, a, false, bQ[kc],
                                                         (short)0, cy, false, false);
            cn = __builtin_amdgcn_wmma_f32_16x16x32_bf16(false, a, false, bA[kc],
                                                         (short)0, cn, false, false);
        }

        // y_t = tanh(...)  ->  out[b][t][u]
#pragma unroll
        for (int k = 0; k < 8; ++k) {
            int M = k + (hi ? 8 : 0);                 // batch row
            out[((size_t)M * TT + t) * DD + ncol] = tanhf(cy[k]);
        }
        __syncthreads();                              // all reads done before next write
        c = cn;
    }
}

extern "C" void kernel_launch(void* const* d_in, const int* in_sizes, int n_in,
                              void* d_out, int out_size, void* d_ws, size_t ws_size,
                              hipStream_t stream) {
    const float* x    = (const float*)d_in[0];  // (16, 4096, 128)
    const float* A    = (const float*)d_in[1];  // (256, 256)
    const float* Bm   = (const float*)d_in[2];  // (128, 256)
    const float* Wmix = (const float*)d_in[3];  // (256, 256)
    const float* bmix = (const float*)d_in[4];  // (256,)
    float* out = (float*)d_out;                 // (16, 4096, 256)

    char* ws = (char*)d_ws;
    const size_t xb_bytes = (size_t)TT * BB * DD * sizeof(float);   // 64 MB
    float*  xB  = (float*)ws;
    __bf16* At  = (__bf16*)(ws + xb_bytes);                         // 128 KB
    __bf16* QWt = (__bf16*)(ws + xb_bytes + (size_t)DD * DD * 2);   // 128 KB
    __bf16* Bt  = (__bf16*)(ws + xb_bytes + (size_t)DD * DD * 4);   //  64 KB

    lmu_prep_transpose<<<256, 256, 0, stream>>>(A, Bm, At, Bt);
    lmu_prep_qw<<<256, 256, 0, stream>>>(Wmix, QWt);
    lmu_xb_gemm<<<(TT / 16) * BB * (DD / 16) / 8, 256, 0, stream>>>(x, Bt, xB);
    lmu_scan<<<1, 512, 0, stream>>>(xB, At, QWt, bmix, out);
}
